// _FEDformerNet_5463198400719
// MI455X (gfx1250) — matrix operand order repaired and checked by reading, not compiled, not run on hardware
//
#include <hip/hip_runtime.h>
#include <hip/hip_bf16.h>
#include <math.h>

// ---------------------------------------------------------------------------
// FEDformer forward for MI455X (gfx1250, wave32, WMMA).
// All heavy math (FFN, forward/inverse truncated DFT, embeddings, projection)
// runs through a templated bf16 WMMA GEMM kernel (v_wmma_f32_16x16x32_bf16)
// with double-buffered LDS tiles and vectorized global loads.
// ---------------------------------------------------------------------------

typedef __attribute__((ext_vector_type(16))) __bf16 v16bf;
typedef __attribute__((ext_vector_type(8)))  __bf16 v8bf;
typedef __attribute__((ext_vector_type(8)))  float  v8f;

__device__ __forceinline__ unsigned int f2bf(float f) {
  unsigned int u = __float_as_uint(f);
  u += 0x7FFFu + ((u >> 16) & 1u);   // round to nearest even
  return u >> 16;
}
__device__ __forceinline__ unsigned int pack2bf(float a, float b) {
  return f2bf(a) | (f2bf(b) << 16);
}

#define TM 128
#define TN 128
#define TK 32
#define LDSW 40   // ushorts per row: 80 bytes -> 16B aligned chunks, bank-spread

// C[M,N] = act(A*Bt^T + bias) + residual.
// MODEA/MODEC: 0 = row-major; 1 = "(b,l,d)" scatter: addr = (m>>9)*2^18 + idx*512 + (m&511)
// ACT: 0 none, 1 exact GELU. Bt is N x K row-major (pre-transposed weights).
template <int MODEA, int MODEC, int ACT>
__global__ __launch_bounds__(256)
void gemm_bf16_kernel(const float* __restrict__ A, const float* __restrict__ Bt,
                      float* __restrict__ C, const float* __restrict__ bias,
                      const float* __restrict__ res, int M, int N, int K)
{
  __shared__ __align__(16) unsigned short sA[2][TM * LDSW];
  __shared__ __align__(16) unsigned short sB[2][TN * LDSW];

  const int tid  = threadIdx.x;
  const int lane = tid & 31;
  const int wave = tid >> 5;          // 0..7 (8 waves, wave32)
  const int wm   = wave & 1;          // 2 wave-rows  x 64
  const int wn   = wave >> 1;         // 4 wave-cols  x 32
  const int h    = lane >> 4;         // lane half
  const int l15  = lane & 15;

  const int m_block = blockIdx.x * TM;
  const int n_block = blockIdx.y * TN;

  v8f acc[4][2];
#pragma unroll
  for (int i = 0; i < 4; ++i)
#pragma unroll
    for (int j = 0; j < 2; ++j) acc[i][j] = {};

  // ---- tile stager: f32 global -> bf16 LDS, float4 loads both modes ----
  auto load_tiles = [&](int buf, int k0) {
    if (MODEA == 0) {
      // k-contiguous: float4 along k, ds_store_b64 packed bf16
#pragma unroll
      for (int it = 0; it < (TM * TK / 4) / 256; ++it) {
        int i = tid + it * 256;
        int m = i >> 3, k4 = (i & 7) * 4;
        const float4 v =
            *reinterpret_cast<const float4*>(&A[(size_t)(m_block + m) * (size_t)K + k0 + k4]);
        uint2 pk = make_uint2(pack2bf(v.x, v.y), pack2bf(v.z, v.w));
        *reinterpret_cast<uint2*>(&sA[buf][m * LDSW + k4]) = pk;
      }
    } else {
      // m-contiguous: float4 along m, scatter 4 rows in LDS
#pragma unroll
      for (int it = 0; it < (TM * TK / 4) / 256; ++it) {
        int i = tid + it * 256;
        int m4 = (i & 31) * 4;      // 0,4,...,124
        int k  = i >> 5;            // 0..31
        int gm = m_block + m4;
        const float4 v = *reinterpret_cast<const float4*>(
            &A[((size_t)(gm >> 9) << 18) + ((size_t)(k0 + k) << 9) + (gm & 511)]);
        sA[buf][(m4 + 0) * LDSW + k] = (unsigned short)f2bf(v.x);
        sA[buf][(m4 + 1) * LDSW + k] = (unsigned short)f2bf(v.y);
        sA[buf][(m4 + 2) * LDSW + k] = (unsigned short)f2bf(v.z);
        sA[buf][(m4 + 3) * LDSW + k] = (unsigned short)f2bf(v.w);
      }
    }
    // B tile (Bt is N x K row-major), guarded on N
#pragma unroll
    for (int it = 0; it < (TN * TK / 4) / 256; ++it) {
      int i = tid + it * 256;
      int n = i >> 3, k4 = (i & 7) * 4;
      int gn = n_block + n;
      float4 v = make_float4(0.f, 0.f, 0.f, 0.f);
      if (gn < N)
        v = *reinterpret_cast<const float4*>(&Bt[(size_t)gn * (size_t)K + k0 + k4]);
      uint2 pk = make_uint2(pack2bf(v.x, v.y), pack2bf(v.z, v.w));
      *reinterpret_cast<uint2*>(&sB[buf][n * LDSW + k4]) = pk;
    }
  };

  const int nk = K / TK;
  load_tiles(0, 0);
  // prefetch tile 1 cachelines (global_prefetch_b8)
  if (nk > 1) {
    int n = tid >> 1, k = (tid & 1) * 16;
    if (n_block + n < N) __builtin_prefetch(&Bt[(size_t)(n_block + n) * (size_t)K + TK + k], 0, 1);
  }
  __syncthreads();

  for (int kt = 0; kt < nk; ++kt) {
    const int cur = kt & 1;
    if (kt + 1 < nk) load_tiles(cur ^ 1, (kt + 1) * TK);

    // ---- B fragments: 16 contiguous bf16 at element offset 16*h ----
    union BF { v16bf v; v8bf p[2]; };
    BF bf[2];
#pragma unroll
    for (int nt = 0; nt < 2; ++nt) {
      const unsigned short* pb = &sB[cur][(wn * 32 + nt * 16 + l15) * LDSW + 16 * h];
      bf[nt].p[0] = *reinterpret_cast<const v8bf*>(pb);
      bf[nt].p[1] = *reinterpret_cast<const v8bf*>(pb + 8);
    }
    // ---- A fragments: 16 contiguous bf16 at element offset 16*h ----
#pragma unroll
    for (int mt = 0; mt < 4; ++mt) {
      const unsigned short* pa = &sA[cur][(wm * 64 + mt * 16 + l15) * LDSW];
      union AF { v16bf v; v8bf p[2]; } af;
      af.p[0] = *reinterpret_cast<const v8bf*>(pa + 16 * h);
      af.p[1] = *reinterpret_cast<const v8bf*>(pa + 16 * h + 8);
#pragma unroll
      for (int nt = 0; nt < 2; ++nt) {
        acc[mt][nt] = __builtin_amdgcn_wmma_f32_16x16x32_bf16(
            false, af.v, false, bf[nt].v, (short)0, acc[mt][nt], false, false);
      }
    }
    __syncthreads();
  }

  // ---- epilogue: C layout lane=(n), vgpr r -> m = 8h + r ----
#pragma unroll
  for (int mt = 0; mt < 4; ++mt) {
#pragma unroll
    for (int nt = 0; nt < 2; ++nt) {
      int gn = n_block + wn * 32 + nt * 16 + l15;
      if (gn >= N) continue;
      float bv = bias ? bias[gn] : 0.0f;
#pragma unroll
      for (int r = 0; r < 8; ++r) {
        int gm = m_block + wm * 64 + mt * 16 + 8 * h + r;
        float v = acc[mt][nt][r] + bv;
        if (ACT == 1) v = 0.5f * v * (1.0f + erff(v * 0.70710678118654752f));
        size_t off;
        if (MODEC == 0) off = (size_t)gm * (size_t)N + gn;
        else            off = ((size_t)(gm >> 9) << 18) + ((size_t)gn << 9) + (gm & 511);
        if (res) v += res[off];
        C[off] = v;
      }
    }
  }
}

template <int MA, int MC, int AC>
static void launch_gemm(const float* A, const float* Bt, float* C, const float* bias,
                        const float* res, int M, int N, int K, hipStream_t s) {
  dim3 g(M / 128, (N + 127) / 128);
  gemm_bf16_kernel<MA, MC, AC><<<g, 256, 0, s>>>(A, Bt, C, bias, res, M, N, K);
}

// out[c][r] = in[r][c]   (in: R x Ccols)
__global__ void transpose_kernel(const float* __restrict__ in, float* __restrict__ out,
                                 int R, int Ccols) {
  int c = blockIdx.x * 32 + (threadIdx.x & 31);
  int r = blockIdx.y * 8 + (threadIdx.x >> 5);
  if (r < R && c < Ccols) out[(size_t)c * R + r] = in[(size_t)r * Ccols + c];
}

// Truncated DFT bases (K=64 modes, L=512), laid out as Bt (N x K) for the GEMM.
__global__ void init_basis_kernel(float* __restrict__ basisF, float* __restrict__ basisI) {
  int i = blockIdx.x * 256 + threadIdx.x;
  if (i < 128 * 512) {          // forward: rows n(0..127)=[re|im] mode, cols k=l
    int n = i >> 9, l = i & 511;
    int mode = n & 63;
    float th = 6.2831853071795864f * (float)((mode * l) & 511) / 512.0f;
    basisF[i] = (n < 64) ? cosf(th) : -sinf(th);
  }
  if (i < 512 * 128) {          // inverse: rows l, cols k(0..127)=[re|im]
    int l = i >> 7, k = i & 127;
    int mode = k & 63;
    float th = 6.2831853071795864f * (float)((mode * l) & 511) / 512.0f;
    float s = (mode == 0) ? 1.0f : 2.0f;
    float v = (k < 64) ? s * cosf(th) : ((mode == 0) ? 0.0f : -s * sinf(th));
    basisI[i] = v * (1.0f / 512.0f);
  }
}

// out[b,l,:] += x_mark @ Wtime + btime + pos[l]
__global__ void timepos_kernel(float* __restrict__ out, const float* __restrict__ xm,
                               const float* __restrict__ Wt, const float* __restrict__ bt,
                               const float* __restrict__ pos) {
  int i = blockIdx.x * 256 + threadIdx.x;
  if (i >= 16384 * 512) return;
  int d = i & 511;
  int bl = i >> 9;
  int l = bl & 511;
  const float* x = xm + (size_t)bl * 4;
  float v = bt[d] + pos[l * 512 + d];
#pragma unroll
  for (int j = 0; j < 4; ++j) v += x[j] * Wt[j * 512 + d];
  out[i] += v;
}

// LayerNorm over last dim (512); one wave32 per row, 8 rows per block.
__global__ __launch_bounds__(256)
void layernorm_kernel(const float* __restrict__ in, float* __restrict__ out,
                      const float* __restrict__ g, const float* __restrict__ b, int rows) {
  int row = blockIdx.x * 8 + (threadIdx.x >> 5);
  int lane = threadIdx.x & 31;
  if (row >= rows) return;
  const float* x = in + (size_t)row * 512;
  float v[16], s = 0.0f, s2 = 0.0f;
#pragma unroll
  for (int j = 0; j < 16; ++j) { v[j] = x[lane + j * 32]; s += v[j]; s2 += v[j] * v[j]; }
#pragma unroll
  for (int o = 16; o > 0; o >>= 1) { s += __shfl_xor(s, o, 32); s2 += __shfl_xor(s2, o, 32); }
  float mean = s * (1.0f / 512.0f);
  float var  = s2 * (1.0f / 512.0f) - mean * mean;
  float inv  = rsqrtf(var + 1e-5f);
  float* y = out + (size_t)row * 512;
#pragma unroll
  for (int j = 0; j < 16; ++j) {
    int d = lane + j * 32;
    y[d] = (v[j] - mean) * inv * g[d] + b[d];
  }
}

// out[b,h,o,k] = sum_e xf[b,h,e,k] * w[h,e,o,k]   (complex), one block per (h,k).
// xf/of rows m = b*512 + h*64 + e|o, cols 0..63 = re, 64..127 = im.
__global__ __launch_bounds__(256)
void modemix_kernel(const float* __restrict__ xf, const float* __restrict__ w,
                    float* __restrict__ of) {
  int hk = blockIdx.x;
  int hh = hk >> 6, k = hk & 63;
  __shared__ float sr[2048], si[2048];
  int tid = threadIdx.x;
  for (int i = tid; i < 2048; i += 256) {
    int b = i >> 6, e = i & 63;
    size_t m = (size_t)b * 512 + hh * 64 + e;
    sr[i] = xf[m * 128 + k];
    si[i] = xf[m * 128 + 64 + k];
  }
  __syncthreads();
  for (int i = tid; i < 2048; i += 256) {
    int b = i >> 6, o = i & 63;
    const float* wb = w + (size_t)hh * 524288 + (size_t)o * 128 + (size_t)k * 2;
    const float* xr = sr + b * 64;
    const float* xi = si + b * 64;
    float ar = 0.0f, ai = 0.0f;
    for (int e = 0; e < 64; ++e) {
      float wr = wb[(size_t)e * 8192];
      float wi = wb[(size_t)e * 8192 + 1];
      ar += xr[e] * wr - xi[e] * wi;
      ai += xr[e] * wi + xi[e] * wr;
    }
    size_t m = (size_t)b * 512 + hh * 64 + o;
    of[m * 128 + k] = ar;
    of[m * 128 + 64 + k] = ai;
  }
}

// score[k] = sum_e Re(q * conj(kf)); softmax over 64 modes; vw = vf * att.
// One block per (b,h), 64 threads (one per mode).  v == k == mem spectrum.
__global__ __launch_bounds__(64)
void crossattn_kernel(const float* __restrict__ qf, const float* __restrict__ kf,
                      float* __restrict__ vwf) {
  int bh = blockIdx.x;
  int b = bh >> 3, hh = bh & 7;
  int k = threadIdx.x;
  __shared__ float ssc[64], sex[64];
  size_t base = ((size_t)b * 512 + hh * 64) * 128;
  float s = 0.0f;
  for (int e = 0; e < 64; ++e) {
    size_t off = base + (size_t)e * 128;
    s += qf[off + k] * kf[off + k] + qf[off + 64 + k] * kf[off + 64 + k];
  }
  ssc[k] = s * 0.125f;   // 1/sqrt(HD=64)
  __syncthreads();
  float mx = -3.4e38f;
  for (int j = 0; j < 64; ++j) mx = fmaxf(mx, ssc[j]);
  float ex = expf(ssc[k] - mx);
  sex[k] = ex;
  __syncthreads();
  float sum = 0.0f;
  for (int j = 0; j < 64; ++j) sum += sex[j];
  float att = ex / sum;
  for (int e = 0; e < 64; ++e) {
    size_t off = base + (size_t)e * 128;
    vwf[off + k]      = kf[off + k] * att;
    vwf[off + 64 + k] = kf[off + 64 + k] * att;
  }
}

// ---------------------------------------------------------------------------

extern "C" void kernel_launch(void* const* d_in, const int* in_sizes, int n_in,
                              void* d_out, int out_size, void* d_ws, size_t ws_size,
                              hipStream_t stream) {
  (void)in_sizes; (void)n_in; (void)out_size; (void)ws_size;

  const float* x_enc  = (const float*)d_in[0];
  const float* xm_enc = (const float*)d_in[1];
  const float* x_dec  = (const float*)d_in[2];
  const float* xm_dec = (const float*)d_in[3];

  int p = 4;
  const float* encValW  = (const float*)d_in[p++];
  const float* encValB  = (const float*)d_in[p++];
  const float* encTimeW = (const float*)d_in[p++];
  const float* encTimeB = (const float*)d_in[p++];
  const float* encPos   = (const float*)d_in[p++];
  const float* decValW  = (const float*)d_in[p++];
  const float* decValB  = (const float*)d_in[p++];
  const float* decTimeW = (const float*)d_in[p++];
  const float* decTimeB = (const float*)d_in[p++];
  const float* decPos   = (const float*)d_in[p++];

  struct EncL { const float *fw, *ln1g, *ln1b, *ln2g, *ln2b, *ff1W, *ff1b, *ff2W, *ff2b; };
  EncL enc[2];
  for (int l = 0; l < 2; ++l) {
    enc[l].fw   = (const float*)d_in[p++];
    enc[l].ln1g = (const float*)d_in[p++];
    enc[l].ln1b = (const float*)d_in[p++];
    enc[l].ln2g = (const float*)d_in[p++];
    enc[l].ln2b = (const float*)d_in[p++];
    enc[l].ff1W = (const float*)d_in[p++];
    enc[l].ff1b = (const float*)d_in[p++];
    enc[l].ff2W = (const float*)d_in[p++];
    enc[l].ff2b = (const float*)d_in[p++];
  }
  const float* fw_self  = (const float*)d_in[p++];
  const float* fw_cross = (const float*)d_in[p++];
  const float* dln1g = (const float*)d_in[p++];
  const float* dln1b = (const float*)d_in[p++];
  const float* dln2g = (const float*)d_in[p++];
  const float* dln2b = (const float*)d_in[p++];
  const float* dln3g = (const float*)d_in[p++];
  const float* dln3b = (const float*)d_in[p++];
  const float* dff1W = (const float*)d_in[p++];
  const float* dff1b = (const float*)d_in[p++];
  const float* dff2W = (const float*)d_in[p++];
  const float* dff2b = (const float*)d_in[p++];
  const float* projW = (const float*)d_in[p++];
  const float* projB = (const float*)d_in[p++];

  // -------- workspace carve-up (floats) --------
  float* ws     = (float*)d_ws;
  float* basisF = ws;                       // 128*512
  float* basisI = basisF + 65536;           // 512*128
  float* wtA    = basisI + 65536;           // up to 2048*512
  float* wtB    = wtA + 1048576;            // up to 512*2048
  float* P0     = wtB + 1048576;            // 16384*512  (enc state / mem)
  float* P1     = P0 + 8388608;
  float* P2     = P1 + 8388608;
  float* P3     = P2 + 8388608;             // normed scratch
  float* hbuf   = P3 + 8388608;             // 16384*2048 FFN hidden
  float* fq1    = hbuf + 33554432;          // 16384*128 spectra
  float* fq2    = fq1 + 2097152;
  float* fqK    = fq2 + 2097152;
  float* fqVW   = fqK + 2097152;

  const int M = 16384;   // B*L = 32*512

  auto transp = [&](const float* W, float* Wt, int R, int Ccols) {
    dim3 g((Ccols + 31) / 32, (R + 7) / 8);
    transpose_kernel<<<g, 256, 0, stream>>>(W, Wt, R, Ccols);
  };
  auto lnorm = [&](const float* in, float* out, const float* g, const float* b) {
    layernorm_kernel<<<M / 8, 256, 0, stream>>>(in, out, g, b, M);
  };

  init_basis_kernel<<<256, 256, 0, stream>>>(basisF, basisI);

  // ===== encoder embedding -> P0 =====
  transp(encValW, wtA, 64, 512);
  launch_gemm<0, 0, 0>(x_enc, wtA, P0, encValB, nullptr, M, 512, 64, stream);
  timepos_kernel<<<(M * 512) / 256, 256, 0, stream>>>(P0, xm_enc, encTimeW, encTimeB, encPos);

  // ===== encoder layers (P0 -> P0) =====
  for (int l = 0; l < 2; ++l) {
    lnorm(P0, P3, enc[l].ln1g, enc[l].ln1b);
    launch_gemm<1, 0, 0>(P3, basisF, fq1, nullptr, nullptr, M, 128, 512, stream); // rfft
    modemix_kernel<<<512, 256, 0, stream>>>(fq1, enc[l].fw, fq2);                 // complex einsum
    launch_gemm<0, 1, 0>(fq2, basisI, P1, nullptr, P0, M, 512, 128, stream);      // irfft + residual
    lnorm(P1, P3, enc[l].ln2g, enc[l].ln2b);
    transp(enc[l].ff1W, wtA, 512, 2048);
    launch_gemm<0, 0, 1>(P3, wtA, hbuf, enc[l].ff1b, nullptr, M, 2048, 512, stream); // FFN1+GELU
    transp(enc[l].ff2W, wtB, 2048, 512);
    launch_gemm<0, 0, 0>(hbuf, wtB, P0, enc[l].ff2b, P1, M, 512, 2048, stream);      // FFN2+res
  }
  // mem = P0

  // ===== decoder embedding -> P1 =====
  transp(decValW, wtA, 64, 512);
  launch_gemm<0, 0, 0>(x_dec, wtA, P1, decValB, nullptr, M, 512, 64, stream);
  timepos_kernel<<<(M * 512) / 256, 256, 0, stream>>>(P1, xm_dec, decTimeW, decTimeB, decPos);

  // ----- decoder self Fourier block: P2 = P1 + fourier(LN(P1)) -----
  lnorm(P1, P3, dln1g, dln1b);
  launch_gemm<1, 0, 0>(P3, basisF, fq1, nullptr, nullptr, M, 128, 512, stream);
  modemix_kernel<<<512, 256, 0, stream>>>(fq1, fw_self, fq2);
  launch_gemm<0, 1, 0>(fq2, basisI, P2, nullptr, P1, M, 512, 128, stream);

  // ----- decoder Fourier cross attention: P1 = P2 + cross(LN(P2), mem) -----
  lnorm(P2, P3, dln2g, dln2b);
  launch_gemm<1, 0, 0>(P3, basisF, fq1, nullptr, nullptr, M, 128, 512, stream);   // qf
  launch_gemm<1, 0, 0>(P0, basisF, fqK, nullptr, nullptr, M, 128, 512, stream);   // kf = vf
  crossattn_kernel<<<256, 64, 0, stream>>>(fq1, fqK, fqVW);
  modemix_kernel<<<512, 256, 0, stream>>>(fqVW, fw_cross, fq2);
  launch_gemm<0, 1, 0>(fq2, basisI, P1, nullptr, P2, M, 512, 128, stream);

  // ----- decoder FFN: P2 = P1 + FFN(LN(P1)) -----
  lnorm(P1, P3, dln3g, dln3b);
  transp(dff1W, wtA, 512, 2048);
  launch_gemm<0, 0, 1>(P3, wtA, hbuf, dff1b, nullptr, M, 2048, 512, stream);
  transp(dff2W, wtB, 2048, 512);
  launch_gemm<0, 0, 0>(hbuf, wtB, P2, dff2b, P1, M, 512, 2048, stream);

  // ===== final projection -> d_out (16384 x 64) =====
  transp(projW, wtA, 512, 64);
  launch_gemm<0, 0, 0>(P2, wtA, (float*)d_out, projB, nullptr, M, 64, 512, stream);
}
